// ToyNet_4758823764759
// MI455X (gfx1250) — compile-verified, matrix operand order checked by reading, not run on hardware
//
#include <hip/hip_runtime.h>
#include <cstdint>
#include <cstddef>

typedef unsigned int u32;
typedef unsigned short u16;
typedef __attribute__((ext_vector_type(16))) __bf16 v16bf;
typedef __attribute__((ext_vector_type(8)))  float  v8f;
typedef __attribute__((ext_vector_type(8)))  u32    v8u;

#define BN_EPS 1e-5f

enum { EPI_F32 = 0, EPI_BN_BF16 = 1, EPI_RELU_BF16 = 2 };

__device__ __forceinline__ u16 f2bf(float f) {
  u32 u = __builtin_bit_cast(u32, f);
  u32 r = (u + 0x7FFFu + ((u >> 16) & 1u)) >> 16;   // round-to-nearest-even
  return (u16)r;
}
__device__ __forceinline__ float bf2f(u16 h) {
  return __builtin_bit_cast(float, (u32)h << 16);
}
// low 32 bits of a generic pointer into shared space == LDS byte offset
__device__ __forceinline__ u32 lds_off(const void* p) {
  return (u32)(uintptr_t)p;
}

// ---------------- small utility kernels ----------------

__global__ void k_cvt_bf16(const float* __restrict__ src, u16* __restrict__ dst, long n) {
  long i = (long)blockIdx.x * blockDim.x + threadIdx.x;
  long st = (long)gridDim.x * blockDim.x;
  for (; i < n; i += st) {
    float v = __builtin_nontemporal_load(&src[i]);
    __builtin_nontemporal_store(f2bf(v), &dst[i]);
  }
}

__global__ void k_zero(float* __restrict__ p, long n) {
  long i = (long)blockIdx.x * blockDim.x + threadIdx.x;
  long st = (long)gridDim.x * blockDim.x;
  for (; i < n; i += st) p[i] = 0.0f;
}

__global__ void k_degree(const int* __restrict__ dst, float* __restrict__ deg, int E) {
  int e = blockIdx.x * blockDim.x + threadIdx.x;
  if (e < E) atomicAdd(&deg[dst[e]], 1.0f);
}

// scatter-add of projected features along edges: s[dst] += pl[src]
__global__ void k_aggregate(const float* __restrict__ pl, const int* __restrict__ src,
                            const int* __restrict__ dst, float* __restrict__ s, int O) {
  int e = blockIdx.x;
  int sr = src[e], dr = dst[e];
  const float* prow = pl + (size_t)sr * O;
  float* srow = s + (size_t)dr * O;
  for (int c = threadIdx.x; c < O; c += blockDim.x) {
    float v = __builtin_nontemporal_load(&prow[c]);
    atomicAdd(&srow[c], v);
  }
}

// out = relu(bn(s/deg + bl + pr)) -> bf16
__global__ void k_finalize(const float* __restrict__ s, const float* __restrict__ pr,
                           const float* __restrict__ deg, const float* __restrict__ bl,
                           const float* __restrict__ g, const float* __restrict__ bt,
                           const float* __restrict__ mn, const float* __restrict__ vr,
                           u16* __restrict__ hout, int N, int O) {
  long total = (long)N * O;
  long i = (long)blockIdx.x * blockDim.x + threadIdx.x;
  long st = (long)gridDim.x * blockDim.x;
  for (; i < total; i += st) {
    int row = (int)(i / O);
    int c = (int)(i - (long)row * O);
    float d = deg[row]; d = d < 1.0f ? 1.0f : d;
    float v = __builtin_nontemporal_load(&s[i]) / d + bl[c]
            + __builtin_nontemporal_load(&pr[i]);
    v = (v - mn[c]) * rsqrtf(vr[c] + BN_EPS) * g[c] + bt[c];
    v = v > 0.0f ? v : 0.0f;
    hout[i] = f2bf(v);
  }
}

// head: [N,64] bf16 -> logits [N,2] + softmax [N,2] (concatenated in d_out)
__global__ void k_head(const u16* __restrict__ h, const float* __restrict__ W3,
                       const float* __restrict__ b3, float* __restrict__ out, int N) {
  int row = blockIdx.x * blockDim.x + threadIdx.x;
  if (row >= N) return;
  const u16* hr = h + (size_t)row * 64;
  float l0 = b3[0], l1 = b3[1];
  #pragma unroll 8
  for (int k = 0; k < 64; ++k) {
    float xv = bf2f(hr[k]);
    l0 += xv * W3[k];
    l1 += xv * W3[64 + k];
  }
  out[(size_t)row * 2 + 0] = l0;
  out[(size_t)row * 2 + 1] = l1;
  float m = l0 > l1 ? l0 : l1;
  float e0 = __expf(l0 - m), e1 = __expf(l1 - m);
  float inv = 1.0f / (e0 + e1);
  out[(size_t)N * 2 + (size_t)row * 2 + 0] = e0 * inv;
  out[(size_t)N * 2 + (size_t)row * 2 + 1] = e1 * inv;
}

// ---------------- WMMA bf16 GEMM: C[M,O] = A[M,K] * W[O,K]^T (+ epilogue) ----------------
// 128x128 block tile, BK=32, 256 threads = 8 waves laid out 4(M) x 2(N);
// each wave computes a 32x64 sub-tile = 2x4 WMMA fragments.
// Double-buffered staging with GLOBAL_LOAD_ASYNC_TO_LDS_B128: tile t+1 streams
// in (ASYNCcnt) while tile t feeds the WMMAs; s_wait_asynccnt 0x4 retires only
// the older tile's 4 per-thread copies. Fragments read back as ds_load_b128
// (row stride = 5 uint4 = 20 words, 16B aligned, bank-conflict free).

__launch_bounds__(256)
__global__ void k_gemm_wmma(const u16* __restrict__ A, int lda,
                            const u16* __restrict__ W, // [O,K] row-major (bf16)
                            int M, int K, int O,
                            float* __restrict__ outF, u16* __restrict__ outH, int ldo,
                            const float* __restrict__ bias,
                            const float* __restrict__ g, const float* __restrict__ bt,
                            const float* __restrict__ mn, const float* __restrict__ vr,
                            int mode) {
  __shared__ uint4 As4[2][128 * 5];  // per buffer: 128 rows x (16 words + 4 pad)
  __shared__ uint4 Bs4[2][128 * 5];

  const int tid  = threadIdx.x;
  const int lane = tid & 31;
  const int wave = tid >> 5;
  const int wm   = wave >> 1;      // 0..3  -> 32-row strip
  const int wn   = wave & 1;       // 0..1  -> 64-col strip
  const int row0 = blockIdx.y * 128;
  const int col0 = blockIdx.x * 128;
  const int khalf = lane >> 4;     // K-half selector (ISA 16-bit fragment layout)
  const int l15   = lane & 15;

  v8f acc[2][4];
  #pragma unroll
  for (int mb = 0; mb < 2; ++mb)
    #pragma unroll
    for (int nb = 0; nb < 4; ++nb) acc[mb][nb] = (v8f){0,0,0,0,0,0,0,0};

  // issue one tile's async copies (4 x b128 per thread: 2 A + 2 B)
  auto issue = [&](int kb, int buf) {
    #pragma unroll
    for (int it = 0; it < 2; ++it) {
      int chunk = it * 256 + tid;        // 512 x 16B chunks per 128x32 tile
      int r  = chunk >> 2;               // tile row / weight row
      int kk = (chunk & 3) * 8;          // bf16 offset within 32-wide K slab
      int ar = row0 + r; if (ar > M - 1) ar = M - 1;
      const u16* ga = A + (size_t)ar * lda + kb + kk;
      u32 la = lds_off(&As4[buf][r * 5 + (kk >> 3)]);
      asm volatile("global_load_async_to_lds_b128 %0, %1, off"
                   :: "v"(la), "v"(ga) : "memory");
      int wc = col0 + r; if (wc > O - 1) wc = O - 1;
      const u16* gb = W + (size_t)wc * K + kb + kk;
      u32 lb = lds_off(&Bs4[buf][r * 5 + (kk >> 3)]);
      asm volatile("global_load_async_to_lds_b128 %0, %1, off"
                   :: "v"(lb), "v"(gb) : "memory");
    }
  };

  const int nk = K >> 5;
  issue(0, 0);
  for (int t = 0; t < nk; ++t) {
    const int buf = t & 1;
    if (t + 1 < nk) {
      issue((t + 1) << 5, buf ^ 1);
      asm volatile("s_wait_asynccnt 0x4" ::: "memory"); // only tile t retired
    } else {
      asm volatile("s_wait_asynccnt 0x0" ::: "memory");
    }
    __syncthreads();

    // ---- A fragments: two 16x32 blocks for this wave's 32 rows ----
    v16bf af[2];
    #pragma unroll
    for (int mb = 0; mb < 2; ++mb) {
      int rowi = wm * 32 + mb * 16 + l15;
      uint4 q0 = As4[buf][rowi * 5 + khalf];       // K 0..7   (or 8..15)
      uint4 q1 = As4[buf][rowi * 5 + 2 + khalf];   // K 16..23 (or 24..31)
      v8u au;
      au[0] = q0.x; au[1] = q0.y; au[2] = q0.z; au[3] = q0.w;
      au[4] = q1.x; au[5] = q1.y; au[6] = q1.z; au[7] = q1.w;
      af[mb] = __builtin_bit_cast(v16bf, au);
    }

    #pragma unroll
    for (int nb = 0; nb < 4; ++nb) {
      int ncol = wn * 64 + nb * 16 + l15;
      uint4 q0 = Bs4[buf][ncol * 5 + khalf];
      uint4 q1 = Bs4[buf][ncol * 5 + 2 + khalf];
      v8u bu;
      bu[0] = q0.x; bu[1] = q0.y; bu[2] = q0.z; bu[3] = q0.w;
      bu[4] = q1.x; bu[5] = q1.y; bu[6] = q1.z; bu[7] = q1.w;
      v16bf bfr = __builtin_bit_cast(v16bf, bu);
      #pragma unroll
      for (int mb = 0; mb < 2; ++mb) {
        acc[mb][nb] = __builtin_amdgcn_wmma_f32_16x16x32_bf16(
            false, af[mb], false, bfr, (short)0, acc[mb][nb], false, false);
      }
    }
    __syncthreads();   // protect buf from being overwritten by tile t+2
  }

  // ---- epilogue: C VGPR r -> row (khalf*8 + r), col = lane&15 ----
  #pragma unroll
  for (int nb = 0; nb < 4; ++nb) {
    int c = col0 + wn * 64 + nb * 16 + l15;
    if (c >= O) continue;
    float bv = 0.0f, gg = 1.0f, bb = 0.0f, mm = 0.0f, iv = 1.0f;
    if (mode != EPI_F32) bv = bias[c];
    if (mode == EPI_BN_BF16) {
      gg = g[c]; bb = bt[c]; mm = mn[c];
      iv = rsqrtf(vr[c] + BN_EPS);
    }
    #pragma unroll
    for (int mb = 0; mb < 2; ++mb) {
      int rbase = row0 + wm * 32 + mb * 16 + khalf * 8;
      #pragma unroll
      for (int r = 0; r < 8; ++r) {
        int rr = rbase + r;
        if (rr >= M) continue;
        float v = acc[mb][nb][r];
        if (mode == EPI_F32) {
          __builtin_nontemporal_store(v, &outF[(size_t)rr * ldo + c]);
        } else {
          v += bv;
          if (mode == EPI_BN_BF16) v = (v - mm) * iv * gg + bb;
          else                     v = v > 0.0f ? v : 0.0f;
          outH[(size_t)rr * ldo + c] = f2bf(v);
        }
      }
    }
  }
}

// ---------------- host ----------------

struct BnP { const float *g, *b, *m, *v; };

static inline void* bump(char*& p, size_t bytes) {
  void* r = p;
  p += (bytes + 255) & ~(size_t)255;
  return r;
}

extern "C" void kernel_launch(void* const* d_in, const int* in_sizes, int n_in,
                              void* d_out, int out_size, void* d_ws, size_t ws_size,
                              hipStream_t stream) {
  (void)in_sizes; (void)n_in; (void)out_size; (void)ws_size;
  const int N = 40000, E = 320000;

  const float* x  = (const float*)d_in[0];
  const int*   ei = (const int*)d_in[1];
  const int* esrc = ei;
  const int* edst = ei + E;

  int pi = 2;
  auto F = [&]() { return (const float*)d_in[pi++]; };
  const float* Wq = F(); const float* bq = F();
  const float* Wc = F(); const float* bc = F();
  const float* Wi = F(); const float* bi = F();
  BnP bnq{F(), F(), F(), F()};
  BnP bnc{F(), F(), F(), F()};
  BnP bni{F(), F(), F(), F()};
  const float *Wl[6], *bl[6], *Wr[6];
  for (int l = 0; l < 6; ++l) { Wl[l] = F(); bl[l] = F(); Wr[l] = F(); }
  BnP bn[6];
  for (int l = 0; l < 6; ++l) bn[l] = BnP{F(), F(), F(), F()};
  const float* W1 = F(); const float* b1 = F();
  const float* W2 = F(); const float* b2 = F();
  const float* W3 = F(); const float* b3 = F();

  // workspace carve-out
  char* wp = (char*)d_ws;
  u16*   xbf = (u16*)bump(wp, (size_t)N * 3584 * 2);
  u16*   hA  = (u16*)bump(wp, (size_t)N * 2048 * 2);
  u16*   hB  = (u16*)bump(wp, (size_t)N * 2048 * 2);
  float* pl  = (float*)bump(wp, (size_t)N * 2048 * 4);
  float* pr  = (float*)bump(wp, (size_t)N * 2048 * 4);
  float* s   = (float*)bump(wp, (size_t)N * 2048 * 4);
  float* deg = (float*)bump(wp, (size_t)N * 4);

  auto cvtW = [&](const float* w, size_t n) -> u16* {
    u16* d = (u16*)bump(wp, n * 2);
    k_cvt_bf16<<<1024, 256, 0, stream>>>(w, d, (long)n);
    return d;
  };
  u16* Wqb = cvtW(Wq, (size_t)512 * 768);
  u16* Wcb = cvtW(Wc, (size_t)512 * 768);
  u16* Wib = cvtW(Wi, (size_t)512 * 2048);
  const int dims[7] = {1536, 2048, 1024, 512, 256, 256, 128};
  u16 *Wlb[6], *Wrb[6];
  for (int l = 0; l < 6; ++l) {
    size_t sz = (size_t)dims[l + 1] * dims[l];
    Wlb[l] = cvtW(Wl[l], sz);
    Wrb[l] = cvtW(Wr[l], sz);
  }
  u16* W1b = cvtW(W1, 128 * 128);
  u16* W2b = cvtW(W2, 64 * 128);

  // degree (structure-only, once per launch)
  k_zero<<<64, 256, 0, stream>>>(deg, N);
  k_degree<<<(E + 255) / 256, 256, 0, stream>>>(edst, deg, E);

  // x -> bf16
  k_cvt_bf16<<<8192, 256, 0, stream>>>(x, xbf, (long)N * 3584);

  auto gemm = [&](const u16* A, int lda, const u16* Wb, int K, int O,
                  float* oF, u16* oH, int ldo, const float* bias,
                  const BnP* bp, int mode) {
    dim3 grid((O + 127) / 128, (N + 127) / 128);
    k_gemm_wmma<<<grid, 256, 0, stream>>>(
        A, lda, Wb, N, K, O, oF, oH, ldo, bias,
        bp ? bp->g : nullptr, bp ? bp->b : nullptr,
        bp ? bp->m : nullptr, bp ? bp->v : nullptr, mode);
  };

  // input projections + BN, concatenated into hA [N,1536] (bf16)
  gemm(xbf + 0,    3584, Wqb, 768,  512, nullptr, hA + 0,    1536, bq, &bnq, EPI_BN_BF16);
  gemm(xbf + 768,  3584, Wcb, 768,  512, nullptr, hA + 512,  1536, bc, &bnc, EPI_BN_BF16);
  gemm(xbf + 1536, 3584, Wib, 2048, 512, nullptr, hA + 1024, 1536, bi, &bni, EPI_BN_BF16);

  // 6 SAGE layers: project-then-aggregate (mean is linear)
  u16* hin = hA;
  u16* hout = hB;
  for (int l = 0; l < 6; ++l) {
    int Kd = dims[l], Od = dims[l + 1];
    gemm(hin, Kd, Wlb[l], Kd, Od, pl, nullptr, Od, nullptr, nullptr, EPI_F32);
    gemm(hin, Kd, Wrb[l], Kd, Od, pr, nullptr, Od, nullptr, nullptr, EPI_F32);
    k_zero<<<8192, 256, 0, stream>>>(s, (long)N * Od);
    k_aggregate<<<E, 256, 0, stream>>>(pl, esrc, edst, s, Od);
    k_finalize<<<8192, 256, 0, stream>>>(s, pr, deg, bl[l],
                                         bn[l].g, bn[l].b, bn[l].m, bn[l].v,
                                         hout, N, Od);
    u16* t = hin; hin = hout; hout = t;
  }

  // head MLP
  u16* t1 = hout;          // ping-pong partner of hin
  gemm(hin, 128, W1b, 128, 128, nullptr, t1, 128, b1, nullptr, EPI_RELU_BF16);
  u16* t2 = (u16*)pl;      // reuse fp32 scratch as bf16 buffer
  gemm(t1, 128, W2b, 128, 64, nullptr, t2, 64, b2, nullptr, EPI_RELU_BF16);
  k_head<<<(N + 255) / 256, 256, 0, stream>>>(t2, W3, b3, (float*)d_out, N);
}
